// AttentionEXTModule_8589934592332
// MI455X (gfx1250) — matrix operand order repaired and checked by reading, not compiled, not run on hardware
//
#include <hip/hip_runtime.h>
#include <hip/hip_bf16.h>

// ---------------------------------------------------------------------------
// MI455X / gfx1250 (CDNA5, wave32) fused implementation.
// Precision: bf16 WMMA (v_wmma_f32_16x16x32_bf16), f32 accumulate.
// N=2048, M=128, D_IN=352, H1=512, D=256, OUT_C=13
// ---------------------------------------------------------------------------

typedef __attribute__((ext_vector_type(16))) __bf16 v16bf;
typedef __attribute__((ext_vector_type(8)))  float  v8f;

union Frag16 { v16bf v; unsigned u[8]; unsigned short h[16]; };
union FragC  { v8f  v; float f[8]; };

static __device__ __forceinline__ unsigned short f2bf(float x) {
  union { float f; unsigned u; } t; t.f = x;
  unsigned r = t.u + 0x7FFFu + ((t.u >> 16) & 1u);   // RNE
  return (unsigned short)(r >> 16);
}
static __device__ __forceinline__ unsigned pack2bf(float a, float b) {
  return (unsigned)f2bf(a) | ((unsigned)f2bf(b) << 16);
}
// 16-bit A 16x32 k order used here (contiguous, not the ISA table order): VGPR v holds K pair; lanes 0-15 get K 0..15,
// lanes 16-31 get K 16..31 (matches B-fragment swizzle).
static __device__ __forceinline__ int amap(int v, int half) {
  return 2 * v + 16 * half;
}

// ---------------- prep kernels ----------------------------------------------
__global__ void k_f32_to_bf16(const float* __restrict__ x,
                              unsigned short* __restrict__ y, int n) {
  int i = blockIdx.x * blockDim.x + threadIdx.x;
  if (i < n) y[i] = f2bf(x[i]);
}

// W is [C][K] row-major f32 (out-features x in-features); emit WMMA B-matrix
// fragments: frag f = kt*(C/16)+nt; per-lane 16 contiguous bf16 (32B).
// Lane l (half=l/16, col=l%16): element e = W[nt*16+col][kt*32 + 16*half + e].
__global__ void k_swizzle_w(const float* __restrict__ W,
                            unsigned short* __restrict__ F, int C, int K) {
  int t = blockIdx.x * blockDim.x + threadIdx.x;
  int total = (K >> 5) * (C >> 4) * 32;
  if (t >= total) return;
  int f = t >> 5, lane = t & 31;
  int ntC = C >> 4;
  int kt = f / ntC, nt = f % ntC;
  int half = lane >> 4, col = lane & 15;
  const float* src = W + (size_t)(nt * 16 + col) * K + kt * 32 + 16 * half;
  unsigned short* dst = F + ((size_t)f * 32 + lane) * 16;
#pragma unroll
  for (int e = 0; e < 16; ++e) dst[e] = f2bf(src[e]);
}

// ---------------- generic fused GEMM + BN(eval) + ReLU ----------------------
// Y[r][c] = relu(g[c]*(sum_k X[r][k]*W[c][k] + b[c]) + be[c])
// X: bf16 [R][K] row-major; Wf: pre-swizzled B-fragments; one 16x16 tile/wave.
__global__ __launch_bounds__(256) void k_gemm_bn_relu(
    const unsigned short* __restrict__ X, const unsigned short* __restrict__ Wf,
    const float* __restrict__ bias, const float* __restrict__ gamma,
    const float* __restrict__ beta, unsigned short* __restrict__ Ybf,
    float* __restrict__ Yf, int R, int C, int K) {
  int lane = threadIdx.x & 31;
  int wave = threadIdx.x >> 5;
  int ntC = C >> 4;
  int nTiles = (R >> 4) * ntC;
  int tile = blockIdx.x * 8 + wave;
  if (tile >= nTiles) return;                 // wave-uniform: EXEC stays full
  int mt = tile / ntC, nt = tile % ntC;
  int half = lane >> 4, r = lane & 15;
  int row0 = mt * 16;

  FragC acc;
#pragma unroll
  for (int v = 0; v < 8; ++v) acc.f[v] = 0.f;

  int nK = K >> 5;
  for (int kt = 0; kt < nK; ++kt) {
    Frag16 fa, fb;
    const unsigned short* xr = X + (size_t)(row0 + r) * K + kt * 32;
#pragma unroll
    for (int v = 0; v < 8; ++v)
      fa.u[v] = *(const unsigned*)(xr + amap(v, half));
    const uint4* bp =
        (const uint4*)(Wf + (((size_t)kt * ntC + nt) * 32 + lane) * 16);
    uint4 q0 = bp[0], q1 = bp[1];
    fb.u[0] = q0.x; fb.u[1] = q0.y; fb.u[2] = q0.z; fb.u[3] = q0.w;
    fb.u[4] = q1.x; fb.u[5] = q1.y; fb.u[6] = q1.z; fb.u[7] = q1.w;
    acc.v = __builtin_amdgcn_wmma_f32_16x16x32_bf16(
        false, fa.v, false, fb.v, (short)0, acc.v, false, false);
  }
  int c = nt * 16 + r;
  float bb = bias[c], gg = gamma[c], ee = beta[c];
#pragma unroll
  for (int v = 0; v < 8; ++v) {
    int row = row0 + v + 8 * half;            // C/D layout (ISA 7.12.2)
    float y = fmaxf(gg * (acc.f[v] + bb) + ee, 0.f);
    if (Ybf) Ybf[(size_t)row * C + c] = f2bf(y);
    if (Yf)  Yf[(size_t)row * C + c] = y;
  }
}

// ---------------- main fused attention kernel -------------------------------
// One workgroup (16 wave32) per n. diff recomputed from Aenc/Benc (L2-hot).
// GEMM1: [128,256]->[128,32]; GEMM2: ->[128,64]; GEMM3: ->[128,256] logits,
// softmax over m (axis M), att[c] = sum_m wei*diff, fc 256->13.
__global__ __launch_bounds__(512) void k_attention_fused(
    const float* __restrict__ Aenc, const float* __restrict__ Benc,
    const unsigned short* __restrict__ w1f, const float* __restrict__ b1,
    const float* __restrict__ g1, const float* __restrict__ e1,
    const unsigned short* __restrict__ w2f, const float* __restrict__ b2,
    const float* __restrict__ g2, const float* __restrict__ e2,
    const unsigned short* __restrict__ w3f, const float* __restrict__ b3,
    const float* __restrict__ g3, const float* __restrict__ e3,
    const float* __restrict__ fcw, const float* __restrict__ fcb,
    float* __restrict__ out) {
  __shared__ unsigned short h1[128 * 40];   // [m][32], stride 40 (pad)
  __shared__ unsigned short h2[128 * 72];   // [m][64], stride 72 (pad)
  __shared__ float att_s[256];

  const int n = blockIdx.x;
  const int lane = threadIdx.x & 31;
  const int wave = threadIdx.x >> 5;        // 0..15
  const int half = lane >> 4, r = lane & 15;
  const float* arow = Aenc + (size_t)n * 256;

  // ---- GEMM1: diff @ mw1^T -> h1   (16 tiles, one per wave; K=256)
  {
    int mt = wave >> 1, nt = wave & 1;
    FragC acc;
#pragma unroll
    for (int v = 0; v < 8; ++v) acc.f[v] = 0.f;
    const float* brow = Benc + (size_t)(mt * 16 + r) * 256;
    for (int kt = 0; kt < 8; ++kt) {
      Frag16 fa, fb;
#pragma unroll
      for (int v = 0; v < 8; ++v) {
        int k = kt * 32 + amap(v, half);
        float2 av = *(const float2*)(arow + k);
        float2 bv = *(const float2*)(brow + k);
        fa.u[v] = pack2bf(av.x - bv.x, av.y - bv.y);
      }
      const uint4* bp =
          (const uint4*)(w1f + (((size_t)kt * 2 + nt) * 32 + lane) * 16);
      uint4 q0 = bp[0], q1 = bp[1];
      fb.u[0] = q0.x; fb.u[1] = q0.y; fb.u[2] = q0.z; fb.u[3] = q0.w;
      fb.u[4] = q1.x; fb.u[5] = q1.y; fb.u[6] = q1.z; fb.u[7] = q1.w;
      acc.v = __builtin_amdgcn_wmma_f32_16x16x32_bf16(
          false, fa.v, false, fb.v, (short)0, acc.v, false, false);
    }
    int c = nt * 16 + r;
    float bb = b1[c], gg = g1[c], ee = e1[c];
#pragma unroll
    for (int v = 0; v < 8; ++v) {
      int m = mt * 16 + v + 8 * half;
      h1[m * 40 + c] = f2bf(fmaxf(gg * (acc.f[v] + bb) + ee, 0.f));
    }
  }
  __syncthreads();

  // ---- GEMM2: h1 @ mw2^T -> h2   (32 tiles, 2 per wave; K=32)
  for (int t = wave; t < 32; t += 16) {
    int mt = t >> 2, nt = t & 3;
    Frag16 fa, fb;
    const unsigned short* xr = h1 + (mt * 16 + r) * 40;
#pragma unroll
    for (int v = 0; v < 8; ++v)
      fa.u[v] = *(const unsigned*)(xr + amap(v, half));
    const uint4* bp = (const uint4*)(w2f + ((size_t)nt * 32 + lane) * 16);
    uint4 q0 = bp[0], q1 = bp[1];
    fb.u[0] = q0.x; fb.u[1] = q0.y; fb.u[2] = q0.z; fb.u[3] = q0.w;
    fb.u[4] = q1.x; fb.u[5] = q1.y; fb.u[6] = q1.z; fb.u[7] = q1.w;
    FragC acc;
#pragma unroll
    for (int v = 0; v < 8; ++v) acc.f[v] = 0.f;
    acc.v = __builtin_amdgcn_wmma_f32_16x16x32_bf16(
        false, fa.v, false, fb.v, (short)0, acc.v, false, false);
    int c = nt * 16 + r;
    float bb = b2[c], gg = g2[c], ee = e2[c];
#pragma unroll
    for (int v = 0; v < 8; ++v) {
      int m = mt * 16 + v + 8 * half;
      h2[m * 72 + c] = f2bf(fmaxf(gg * (acc.f[v] + bb) + ee, 0.f));
    }
  }
  __syncthreads();

  // ---- GEMM3 + softmax(m) + weighted reduce: each wave owns 16 channels
  {
    const int nt = wave;                    // column strip c0 = nt*16
    const int c = nt * 16 + r;
    FragC acc[8];
#pragma unroll
    for (int mt = 0; mt < 8; ++mt)
#pragma unroll
      for (int v = 0; v < 8; ++v) acc[mt].f[v] = 0.f;

    for (int kt = 0; kt < 2; ++kt) {        // K = 64
      Frag16 fb;
      const uint4* bp =
          (const uint4*)(w3f + (((size_t)kt * 16 + nt) * 32 + lane) * 16);
      uint4 q0 = bp[0], q1 = bp[1];
      fb.u[0] = q0.x; fb.u[1] = q0.y; fb.u[2] = q0.z; fb.u[3] = q0.w;
      fb.u[4] = q1.x; fb.u[5] = q1.y; fb.u[6] = q1.z; fb.u[7] = q1.w;
#pragma unroll
      for (int mt = 0; mt < 8; ++mt) {
        Frag16 fa;
        const unsigned short* xr = h2 + (mt * 16 + r) * 72 + kt * 32;
#pragma unroll
        for (int v = 0; v < 8; ++v)
          fa.u[v] = *(const unsigned*)(xr + amap(v, half));
        acc[mt].v = __builtin_amdgcn_wmma_f32_16x16x32_bf16(
            false, fa.v, false, fb.v, (short)0, acc[mt].v, false, false);
      }
    }
    // affine+relu in place; softmax over m: 64 values here + 64 in lane^16
    float bb = b3[c], gg = g3[c], ee = e3[c];
    float mx = 0.f;                          // relu output >= 0
#pragma unroll
    for (int mt = 0; mt < 8; ++mt)
#pragma unroll
      for (int v = 0; v < 8; ++v) {
        float y = fmaxf(gg * (acc[mt].f[v] + bb) + ee, 0.f);
        acc[mt].f[v] = y;
        mx = fmaxf(mx, y);
      }
    mx = fmaxf(mx, __shfl_xor(mx, 16, 32));
    float s = 0.f;
#pragma unroll
    for (int mt = 0; mt < 8; ++mt)
#pragma unroll
      for (int v = 0; v < 8; ++v) {
        float w = __expf(acc[mt].f[v] - mx);
        acc[mt].f[v] = w;
        s += w;
      }
    s += __shfl_xor(s, 16, 32);
    float inv = 1.f / s;
    float ac = arow[c];
    float p = 0.f;
#pragma unroll
    for (int mt = 0; mt < 8; ++mt)
#pragma unroll
      for (int v = 0; v < 8; ++v) {
        int m = mt * 16 + v + 8 * half;
        float d = ac - Benc[(size_t)m * 256 + c];
        p += acc[mt].f[v] * inv * d;
      }
    p += __shfl_xor(p, 16, 32);
    if (half == 0) att_s[c] = p;
  }
  __syncthreads();

  // ---- final FC 256 -> 13
  if (threadIdx.x < 13) {
    int j = threadIdx.x;
    float a = fcb[j];
    const float* wr = fcw + (size_t)j * 256;
    for (int cc = 0; cc < 256; ++cc) a += wr[cc] * att_s[cc];
    out[(size_t)n * 13 + j] = a;
  }
}

// ---------------------------------------------------------------------------
extern "C" void kernel_launch(void* const* d_in, const int* in_sizes, int n_in,
                              void* d_out, int out_size, void* d_ws,
                              size_t ws_size, hipStream_t stream) {
  (void)in_sizes; (void)n_in; (void)out_size; (void)ws_size;
  const float* ext = (const float*)d_in[0];
  const float* lab = (const float*)d_in[1];
  const float* w1a = (const float*)d_in[2];
  const float* b1a = (const float*)d_in[3];
  const float* g1a = (const float*)d_in[4];
  const float* be1a = (const float*)d_in[5];
  const float* w1b = (const float*)d_in[6];
  const float* b1b = (const float*)d_in[7];
  const float* g1b = (const float*)d_in[8];
  const float* be1b = (const float*)d_in[9];
  const float* w2a = (const float*)d_in[10];
  const float* b2a = (const float*)d_in[11];
  const float* g2a = (const float*)d_in[12];
  const float* be2a = (const float*)d_in[13];
  const float* w2b = (const float*)d_in[14];
  const float* b2b = (const float*)d_in[15];
  const float* g2b = (const float*)d_in[16];
  const float* be2b = (const float*)d_in[17];
  const float* mw1 = (const float*)d_in[18];
  const float* mb1 = (const float*)d_in[19];
  const float* mg1 = (const float*)d_in[20];
  const float* mbe1 = (const float*)d_in[21];
  const float* mw2 = (const float*)d_in[22];
  const float* mb2 = (const float*)d_in[23];
  const float* mg2 = (const float*)d_in[24];
  const float* mbe2 = (const float*)d_in[25];
  const float* mw3 = (const float*)d_in[26];
  const float* mb3 = (const float*)d_in[27];
  const float* mg3 = (const float*)d_in[28];
  const float* mbe3 = (const float*)d_in[29];
  const float* fcw = (const float*)d_in[30];
  const float* fcb = (const float*)d_in[31];

  char* base = (char*)d_ws;
  size_t off = 0;
  auto take = [&](size_t bytes) -> void* {
    void* p = base + off;
    off += (bytes + 255) & ~(size_t)255;
    return p;
  };
  unsigned short* Xe  = (unsigned short*)take((size_t)2048 * 352 * 2);
  unsigned short* Xl  = (unsigned short*)take((size_t)128 * 352 * 2);
  unsigned short* F1a = (unsigned short*)take((size_t)352 * 512 * 2);
  unsigned short* F1b = (unsigned short*)take((size_t)512 * 256 * 2);
  unsigned short* F2a = (unsigned short*)take((size_t)352 * 512 * 2);
  unsigned short* F2b = (unsigned short*)take((size_t)512 * 256 * 2);
  unsigned short* Fm1 = (unsigned short*)take((size_t)256 * 32 * 2);
  unsigned short* Fm2 = (unsigned short*)take((size_t)32 * 64 * 2);
  unsigned short* Fm3 = (unsigned short*)take((size_t)64 * 256 * 2);
  unsigned short* H1a = (unsigned short*)take((size_t)2048 * 512 * 2);
  unsigned short* H1b = (unsigned short*)take((size_t)128 * 512 * 2);
  float* Aenc = (float*)take((size_t)2048 * 256 * 4);
  float* Benc = (float*)take((size_t)128 * 256 * 4);

  // activations -> bf16
  int ne = 2048 * 352;
  k_f32_to_bf16<<<(ne + 255) / 256, 256, 0, stream>>>(ext, Xe, ne);
  int nl = 128 * 352;
  k_f32_to_bf16<<<(nl + 255) / 256, 256, 0, stream>>>(lab, Xl, nl);

  // weight fragment swizzles (threads = K*C/16)
  auto swz = [&](const float* W, unsigned short* F, int C, int K) {
    int t = K * C / 16;
    k_swizzle_w<<<(t + 255) / 256, 256, 0, stream>>>(W, F, C, K);
  };
  swz(w1a, F1a, 512, 352);
  swz(w1b, F1b, 256, 512);
  swz(w2a, F2a, 512, 352);
  swz(w2b, F2b, 256, 512);
  swz(mw1, Fm1, 32, 256);
  swz(mw2, Fm2, 64, 32);
  swz(mw3, Fm3, 256, 64);

  // encoders (tiles = (R/16)*(C/16), 8 tiles/block)
  k_gemm_bn_relu<<<(128 * 32 + 7) / 8, 256, 0, stream>>>(
      Xe, F1a, b1a, g1a, be1a, H1a, nullptr, 2048, 512, 352);
  k_gemm_bn_relu<<<(8 * 32 + 7) / 8, 256, 0, stream>>>(
      Xl, F2a, b2a, g2a, be2a, H1b, nullptr, 128, 512, 352);
  k_gemm_bn_relu<<<(128 * 16 + 7) / 8, 256, 0, stream>>>(
      H1a, F1b, b1b, g1b, be1b, nullptr, Aenc, 2048, 256, 512);
  k_gemm_bn_relu<<<(8 * 16 + 7) / 8, 256, 0, stream>>>(
      H1b, F2b, b2b, g2b, be2b, nullptr, Benc, 128, 256, 512);

  // fused attention: one workgroup per n
  k_attention_fused<<<2048, 512, 0, stream>>>(
      Aenc, Benc, Fm1, mb1, mg1, mbe1, Fm2, mb2, mg2, mbe2, Fm3, mb3, mg3,
      mbe3, fcw, fcb, (float*)d_out);
}
